// GGCN_EDGE_55241869361501
// MI455X (gfx1250) — matrix operand order repaired and checked by reading, not compiled, not run on hardware
//
#include <hip/hip_runtime.h>
#include <hip/hip_bf16.h>

// ---------------- problem constants ----------------
#define NN      20000     // nodes   (multiple of 16)
#define NE      320000    // edges   (multiple of 16)
#define NG      64        // graphs
#define NODE_F  100
#define EDGE_F  50
#define DIMF    50
#define NL      5
#define INV_SQRT_D 0.14142135623730950488f   // 1/sqrt(50)

// WMMA vector types (probe-confirmed signatures)
typedef __attribute__((ext_vector_type(16))) __bf16 v16bf;
typedef __attribute__((ext_vector_type(8)))  float  v8f;

union BF16x16 { v16bf v; unsigned short s[16]; };

__device__ __forceinline__ unsigned short f32_to_bf16(float f) {
    union { float f; unsigned int u; } x; x.f = f;
    unsigned int u = x.u;
    unsigned int r = u + 0x7FFFu + ((u >> 16) & 1u);   // round-to-nearest-even
    return (unsigned short)(r >> 16);
}

// ---------------- fragment geometry ----------------
// B fragment (32x16 K-by-N, bf16): lane%16 = N, 16 packed bf16 = K in the same k order as A.
// Stored as 512 ushorts per fragment: frag[lane*16 + j].
// A fragment (16x32 M-by-K, bf16): lanes 0-15 = rows M (K 0..7 & 16..23),
// lanes 16-31 = same rows (K 8..15 & 24..31).

#define FRAG_USH 512

// fragment region offsets (units of fragments)
#define FR_PREN  0            // 4 ktiles * 4 ntiles
#define FR_PREE  16           // 2*4
#define FR_GATE  24           // per layer 5*4 = 20  -> 100
#define FR_SRC   124          // per layer 2*4 = 8   -> 40
#define FR_DST   164          // per layer 8         -> 40
#define FR_TOTAL 204

// stats buffer layout (floats)
#define S_SUM_N  0
#define S_SQ_N   64
#define S_SUM_E  128
#define S_SQ_E   192
#define S_SC1N   256
#define S_SH1N   320
#define S_SC1E   384
#define S_SH1E   448
#define S_SUM2N  512
#define S_SQ2N   576
#define S_SUM2E  640
#define S_SQ2E   704
#define S_SC2N   768
#define S_SH2N   832
#define S_SC2E   896
#define S_SH2E   960
#define S_TOTAL  1024

// ---------------- weight fragment prep ----------------
__global__ void prep_frags(const float* __restrict__ W, int K, int Ncols,
                           unsigned short* __restrict__ out) {
    const int frag = blockIdx.x;           // kt*4 + nt
    const int kt = frag >> 2, nt = frag & 3;
    const int lane = threadIdx.x & 31;
    const int n = nt * 16 + (lane & 15);
    const int kh = (lane & 16) ? 8 : 0;
    unsigned short* o = out + frag * FRAG_USH + lane * 16;
#pragma unroll
    for (int j = 0; j < 16; ++j) {
        int k = kt * 32 + ((j < 8) ? (kh + j) : (16 + kh + (j - 8)));
        float v = (k < K && n < Ncols) ? W[k * Ncols + n] : 0.0f;
        o[j] = f32_to_bf16(v);
    }
}

__device__ __forceinline__ v16bf load_bfrag(const unsigned short* __restrict__ base,
                                            int fragIdx, int lane) {
    return *(const v16bf*)(base + fragIdx * FRAG_USH + lane * 16);
}

// A fragment from an f32 row (pre-projection inputs); rows always valid.
__device__ __forceinline__ v16bf build_a_f32(const float* __restrict__ rowp,
                                             int K, int kt, int lane) {
    BF16x16 a;
    const int kh = (lane & 16) ? 8 : 0;
#pragma unroll
    for (int j = 0; j < 8; ++j) {
        int k0 = kt * 32 + kh + j;
        int k1 = kt * 32 + 16 + kh + j;
        a.s[j]     = f32_to_bf16((k0 < K) ? rowp[k0] : 0.0f);
        a.s[8 + j] = f32_to_bf16((k1 < K) ? rowp[k1] : 0.0f);
    }
    return a.v;
}

// A fragment from a bf16 mirror row (no conversion, no row guard)
__device__ __forceinline__ v16bf build_a_bf(const unsigned short* __restrict__ rowp,
                                            int K, int kt, int lane) {
    BF16x16 a;
    const int kh = (lane & 16) ? 8 : 0;
#pragma unroll
    for (int j = 0; j < 8; ++j) {
        int k0 = kt * 32 + kh + j;
        int k1 = kt * 32 + 16 + kh + j;
        a.s[j]     = (k0 < K) ? rowp[k0] : (unsigned short)0;
        a.s[8 + j] = (k1 < K) ? rowp[k1] : (unsigned short)0;
    }
    return a.v;
}

// ---------------- pre-projections: f32 in -> f32 out + bf16 mirror ---------
template <int KT>
__global__ void wmma_rowgemm_relu(const float* __restrict__ A, int rows, int K,
                                  const unsigned short* __restrict__ fragbase,
                                  const float* __restrict__ bias,
                                  float* __restrict__ out,
                                  unsigned short* __restrict__ outbf) {
    const int wave = threadIdx.x >> 5;
    const int lane = threadIdx.x & 31;
    const int tile = blockIdx.x * (blockDim.x >> 5) + wave;
    const int rowbase = tile * 16;
    if (rowbase >= rows) return;           // rows % 16 == 0: tiles fully valid

    const int r = rowbase + (lane & 15);
    const float* rowp = A + r * K;

    v16bf afrag[KT];
#pragma unroll
    for (int kt = 0; kt < KT; ++kt) afrag[kt] = build_a_f32(rowp, K, kt, lane);

#pragma unroll
    for (int nt = 0; nt < 4; ++nt) {
        const int n = nt * 16 + (lane & 15);
        const float bb = (n < DIMF) ? bias[n] : 0.0f;
        v8f acc = {bb, bb, bb, bb, bb, bb, bb, bb};
#pragma unroll
        for (int kt = 0; kt < KT; ++kt) {
            v16bf b = load_bfrag(fragbase, kt * 4 + nt, lane);
            acc = __builtin_amdgcn_wmma_f32_16x16x32_bf16(false, afrag[kt],
                                                          false, b,
                                                          (short)0, acc, false, false);
        }
        if (n < DIMF) {
            const int mbase = (lane & 16) ? 8 : 0;
#pragma unroll
            for (int rr = 0; rr < 8; ++rr) {
                int row = rowbase + mbase + rr;
                float v = fmaxf(acc[rr], 0.0f);
                out[row * DIMF + n] = v;
                outbf[row * DIMF + n] = f32_to_bf16(v);
            }
        }
    }
}

// ---------------- node linears: dst_feat and h(=src_feat) -----------------
__global__ void node_linear(const unsigned short* __restrict__ xbf,
                            const unsigned short* __restrict__ srcFrag,
                            const unsigned short* __restrict__ dstFrag,
                            const float* __restrict__ src_b,
                            const float* __restrict__ dst_b,
                            float* __restrict__ hbuf,
                            float* __restrict__ dstf) {
    const int wave = threadIdx.x >> 5;
    const int lane = threadIdx.x & 31;
    const int tile = blockIdx.x * (blockDim.x >> 5) + wave;
    const int rowbase = tile * 16;
    if (rowbase >= NN) return;

    const int r = rowbase + (lane & 15);
    const unsigned short* rowp = xbf + r * DIMF;

    v16bf afrag[2];
#pragma unroll
    for (int kt = 0; kt < 2; ++kt) afrag[kt] = build_a_bf(rowp, DIMF, kt, lane);

#pragma unroll
    for (int nt = 0; nt < 4; ++nt) {
        const int n = nt * 16 + (lane & 15);
        const float sb = (n < DIMF) ? src_b[n] : 0.0f;
        const float db = (n < DIMF) ? dst_b[n] : 0.0f;
        v8f accS = {sb, sb, sb, sb, sb, sb, sb, sb};
        v8f accD = {db, db, db, db, db, db, db, db};
#pragma unroll
        for (int kt = 0; kt < 2; ++kt) {
            v16bf bs = load_bfrag(srcFrag, kt * 4 + nt, lane);
            v16bf bd = load_bfrag(dstFrag, kt * 4 + nt, lane);
            accS = __builtin_amdgcn_wmma_f32_16x16x32_bf16(false, afrag[kt], false, bs,
                                                           (short)0, accS, false, false);
            accD = __builtin_amdgcn_wmma_f32_16x16x32_bf16(false, afrag[kt], false, bd,
                                                           (short)0, accD, false, false);
        }
        if (n < DIMF) {
            const int mbase = (lane & 16) ? 8 : 0;
#pragma unroll
            for (int rr = 0; rr < 8; ++rr) {
                int row = rowbase + mbase + rr;
                hbuf[row * DIMF + n] = accS[rr];
                dstf[row * DIMF + n] = accD[rr];
            }
        }
    }
}

// ---------------- per-edge gate GEMM + sigmoid gather/scatter --------------
__device__ __forceinline__ unsigned short fetch_cat_bf(
        const unsigned short* __restrict__ xbf,
        const unsigned short* __restrict__ ebf,
        int sI, int dI, int e, int k) {
    if (k < 50)  return xbf[sI * DIMF + k];
    if (k < 100) return xbf[dI * DIMF + (k - 50)];
    if (k < 150) return ebf[e * DIMF + (k - 100)];
    return (unsigned short)0;
}

__global__ void edge_gate(const unsigned short* __restrict__ xbf,
                          const unsigned short* __restrict__ ebf,
                          const int* __restrict__ ei_src,
                          const int* __restrict__ ei_dst,
                          const unsigned short* __restrict__ gateFrag,
                          const float* __restrict__ gate_b,
                          const float* __restrict__ dstf,
                          float* __restrict__ hbuf,
                          float* __restrict__ mbuf,
                          float* __restrict__ stats) {
    __shared__ float tileLds[8][16][68];
    __shared__ int sIdx[8][16], dIdx[8][16];
    __shared__ float esum[64], esq[64];

    const int wave = threadIdx.x >> 5;
    const int lane = threadIdx.x & 31;
    const int tile = blockIdx.x * 8 + wave;      // NE/16 = 20000 tiles exactly
    const int rowbase = tile * 16;

    if (threadIdx.x < 64) { esum[threadIdx.x] = 0.0f; esq[threadIdx.x] = 0.0f; }
    if (lane < 16) {
        int e = rowbase + lane;
        sIdx[wave][lane] = ei_src[e];
        dIdx[wave][lane] = ei_dst[e];
    }
    __syncthreads();

    const int myRow = lane & 15;
    const int e0 = rowbase + myRow;
    const int sI = sIdx[wave][myRow];
    const int dI = dIdx[wave][myRow];
    const int kh = (lane & 16) ? 8 : 0;

    v16bf afrag[5];
#pragma unroll
    for (int kt = 0; kt < 5; ++kt) {
        BF16x16 a;
#pragma unroll
        for (int j = 0; j < 8; ++j) {
            int k0 = kt * 32 + kh + j;
            int k1 = kt * 32 + 16 + kh + j;
            a.s[j]     = fetch_cat_bf(xbf, ebf, sI, dI, e0, k0);
            a.s[8 + j] = fetch_cat_bf(xbf, ebf, sI, dI, e0, k1);
        }
        afrag[kt] = a.v;
    }

#pragma unroll
    for (int nt = 0; nt < 4; ++nt) {
        const int n = nt * 16 + (lane & 15);
        const float gb = (n < DIMF) ? gate_b[n] : 0.0f;
        v8f acc = {gb, gb, gb, gb, gb, gb, gb, gb};
#pragma unroll
        for (int kt = 0; kt < 5; ++kt) {
            v16bf b = load_bfrag(gateFrag, kt * 4 + nt, lane);
            acc = __builtin_amdgcn_wmma_f32_16x16x32_bf16(false, afrag[kt], false, b,
                                                          (short)0, acc, false, false);
        }
        const int mbase = (lane & 16) ? 8 : 0;
#pragma unroll
        for (int rr = 0; rr < 8; ++rr) tileLds[wave][mbase + rr][n] = acc[rr];
    }
    __syncthreads();

    // row-wise epilogue: sigmoid gate, scatter-add, edge BN moments
    const int fA = lane;
    const int fB = lane + 32;
    float sA = 0.f, qA = 0.f, sB = 0.f, qB = 0.f;
#pragma unroll 4
    for (int rr = 0; rr < 16; ++rr) {
        const int e = rowbase + rr;
        const int s = sIdx[wave][rr];
        const int d = dIdx[wave][rr];
        if (fA < DIMF) {
            float mv = tileLds[wave][rr][fA];
            mbuf[e * DIMF + fA] = mv;
            float sg = 1.0f / (1.0f + __expf(-mv * INV_SQRT_D));
            atomicAdd(&hbuf[s * DIMF + fA], sg * dstf[d * DIMF + fA]);
            sA += mv; qA += mv * mv;
        }
        if (fB < DIMF) {
            float mv = tileLds[wave][rr][fB];
            mbuf[e * DIMF + fB] = mv;
            float sg = 1.0f / (1.0f + __expf(-mv * INV_SQRT_D));
            atomicAdd(&hbuf[s * DIMF + fB], sg * dstf[d * DIMF + fB]);
            sB += mv; qB += mv * mv;
        }
    }
    if (fA < DIMF) { atomicAdd(&esum[fA], sA); atomicAdd(&esq[fA], qA); }
    if (fB < DIMF) { atomicAdd(&esum[fB], sB); atomicAdd(&esq[fB], qB); }
    __syncthreads();
    if (threadIdx.x < DIMF) {
        atomicAdd(&stats[S_SUM_E + threadIdx.x], esum[threadIdx.x]);
        atomicAdd(&stats[S_SQ_E  + threadIdx.x], esq[threadIdx.x]);
    }
}

// ---------------- column (per-feature) moment reduction --------------------
__global__ void col_stats(const float* __restrict__ buf, int total,
                          float* __restrict__ sumOut, float* __restrict__ sqOut) {
    __shared__ float ls[64], lq[64];
    if (threadIdx.x < 64) { ls[threadIdx.x] = 0.0f; lq[threadIdx.x] = 0.0f; }
    __syncthreads();
    for (int idx = blockIdx.x * blockDim.x + threadIdx.x; idx < total;
         idx += gridDim.x * blockDim.x) {
        float v = buf[idx];
        int f = idx % DIMF;
        atomicAdd(&ls[f], v);
        atomicAdd(&lq[f], v * v);
    }
    __syncthreads();
    if (threadIdx.x < DIMF) {
        atomicAdd(&sumOut[threadIdx.x], ls[threadIdx.x]);
        atomicAdd(&sqOut[threadIdx.x],  lq[threadIdx.x]);
    }
}

__global__ void bn_finalize(const float* __restrict__ sum, const float* __restrict__ sq,
                            float invCnt, const float* __restrict__ g,
                            const float* __restrict__ b,
                            float* __restrict__ scale, float* __restrict__ shift) {
    int f = threadIdx.x;
    if (f < DIMF) {
        float mu  = sum[f] * invCnt;
        float var = sq[f] * invCnt - mu * mu;
        float rs  = rsqrtf(var + 1e-5f);
        float sc  = g[f] * rs;
        scale[f] = sc;
        shift[f] = b[f] - mu * sc;
    }
}

// in-place bn1 apply + relu + bn2 moments
__global__ void bn_apply_relu_stats(float* __restrict__ buf, int total,
                                    const float* __restrict__ scale,
                                    const float* __restrict__ shift,
                                    float* __restrict__ sumOut, float* __restrict__ sqOut) {
    __shared__ float ls[64], lq[64];
    if (threadIdx.x < 64) { ls[threadIdx.x] = 0.0f; lq[threadIdx.x] = 0.0f; }
    __syncthreads();
    for (int idx = blockIdx.x * blockDim.x + threadIdx.x; idx < total;
         idx += gridDim.x * blockDim.x) {
        int f = idx % DIMF;
        float v = fmaxf(buf[idx] * scale[f] + shift[f], 0.0f);
        buf[idx] = v;
        atomicAdd(&ls[f], v);
        atomicAdd(&lq[f], v * v);
    }
    __syncthreads();
    if (threadIdx.x < DIMF) {
        atomicAdd(&sumOut[threadIdx.x], ls[threadIdx.x]);
        atomicAdd(&sqOut[threadIdx.x],  lq[threadIdx.x]);
    }
}

// out += bn2(nbuf); also refresh bf16 mirror of out
__global__ void bn_residual_mirror(const float* __restrict__ nbuf, int total,
                                   const float* __restrict__ scale,
                                   const float* __restrict__ shift,
                                   float* __restrict__ outbuf,
                                   unsigned short* __restrict__ mirror) {
    for (int idx = blockIdx.x * blockDim.x + threadIdx.x; idx < total;
         idx += gridDim.x * blockDim.x) {
        int f = idx % DIMF;
        float v = outbuf[idx] + nbuf[idx] * scale[f] + shift[f];
        outbuf[idx] = v;
        mirror[idx] = f32_to_bf16(v);
    }
}

__global__ void zero_buf(float* __restrict__ p, int n) {
    for (int i = blockIdx.x * blockDim.x + threadIdx.x; i < n;
         i += gridDim.x * blockDim.x) p[i] = 0.0f;
}

__global__ void pool_kernel(const float* __restrict__ out_x,
                            const int* __restrict__ batch,
                            float* __restrict__ pooled) {
    for (int idx = blockIdx.x * blockDim.x + threadIdx.x; idx < NN * DIMF;
         idx += gridDim.x * blockDim.x) {
        int row = idx / DIMF, f = idx - row * DIMF;
        atomicAdd(&pooled[batch[row] * DIMF + f], out_x[idx]);
    }
}

__global__ void head_kernel(const float* __restrict__ pooled,
                            const float* __restrict__ postW, const float* __restrict__ postb,
                            const float* __restrict__ outW,  const float* __restrict__ outb,
                            float* __restrict__ out) {
    __shared__ float hid[NG * 100];
    for (int t = threadIdx.x; t < NG * 100; t += blockDim.x) {
        int g = t / 100, j = t - g * 100;
        float a = postb[j];
#pragma unroll 5
        for (int k = 0; k < DIMF; ++k) a += pooled[g * DIMF + k] * postW[k * 100 + j];
        hid[t] = fmaxf(a, 0.0f);
    }
    __syncthreads();
    for (int g = threadIdx.x; g < NG; g += blockDim.x) {
        float a = outb[0];
#pragma unroll 4
        for (int j = 0; j < 100; ++j) a += hid[g * 100 + j] * outW[j];
        out[g] = a;
    }
}

// ---------------- host-side orchestration ----------------
extern "C" void kernel_launch(void* const* d_in, const int* in_sizes, int n_in,
                              void* d_out, int out_size, void* d_ws, size_t ws_size,
                              hipStream_t stream) {
    const float* x          = (const float*)d_in[0];
    const float* edge_attr  = (const float*)d_in[1];
    const int*   edge_index = (const int*)d_in[2];       // [2][NE]
    const int*   batch      = (const int*)d_in[3];
    const float* preN_W = (const float*)d_in[4];  const float* preN_b = (const float*)d_in[5];
    const float* preE_W = (const float*)d_in[6];  const float* preE_b = (const float*)d_in[7];
    const float* gate_W = (const float*)d_in[8];  const float* gate_b = (const float*)d_in[9];
    const float* src_W  = (const float*)d_in[10]; const float* src_b  = (const float*)d_in[11];
    const float* dst_W  = (const float*)d_in[12]; const float* dst_b  = (const float*)d_in[13];
    const float* bn_in_n_g  = (const float*)d_in[14]; const float* bn_in_n_b  = (const float*)d_in[15];
    const float* bn_in_e_g  = (const float*)d_in[16]; const float* bn_in_e_b  = (const float*)d_in[17];
    const float* bn_out_n_g = (const float*)d_in[18]; const float* bn_out_n_b = (const float*)d_in[19];
    const float* bn_out_e_g = (const float*)d_in[20]; const float* bn_out_e_b = (const float*)d_in[21];
    const float* post_W = (const float*)d_in[22]; const float* post_b = (const float*)d_in[23];
    const float* out_W  = (const float*)d_in[24]; const float* out_b  = (const float*)d_in[25];

    // workspace carve
    float* ws    = (float*)d_ws;
    float* out_x = ws;                         // NN*DIMF  = 1,000,000 f
    float* dstf  = out_x + NN * DIMF;          // 1,000,000 f
    float* hbuf  = dstf + NN * DIMF;           // 1,000,000 f
    float* out_e = hbuf + NN * DIMF;           // 16,000,000 f
    float* mbuf  = out_e + (long)NE * DIMF;    // 16,000,000 f
    float* stats = mbuf + (long)NE * DIMF;     // 1024 f
    float* pooled= stats + S_TOTAL;            // NG*DIMF f
    unsigned short* wf  = (unsigned short*)(pooled + NG * DIMF);   // 204*512 ush
    unsigned short* xbf = wf + FR_TOTAL * FRAG_USH;                // NN*DIMF ush
    unsigned short* ebf = xbf + NN * DIMF;                         // NE*DIMF ush

    const int* ei_src = edge_index;
    const int* ei_dst = edge_index + NE;

    // ---- weight fragment prep ----
    prep_frags<<<16, 32, 0, stream>>>(preN_W, NODE_F, DIMF, wf + FR_PREN * FRAG_USH);
    prep_frags<<<8,  32, 0, stream>>>(preE_W, EDGE_F, DIMF, wf + FR_PREE * FRAG_USH);
    for (int i = 0; i < NL; ++i) {
        prep_frags<<<20, 32, 0, stream>>>(gate_W + i * 150 * DIMF, 150, DIMF,
                                          wf + (FR_GATE + i * 20) * FRAG_USH);
        prep_frags<<<8, 32, 0, stream>>>(src_W + i * DIMF * DIMF, DIMF, DIMF,
                                         wf + (FR_SRC + i * 8) * FRAG_USH);
        prep_frags<<<8, 32, 0, stream>>>(dst_W + i * DIMF * DIMF, DIMF, DIMF,
                                         wf + (FR_DST + i * 8) * FRAG_USH);
    }

    // ---- pre-projections ----
    wmma_rowgemm_relu<4><<<(NN / 16 + 7) / 8, 256, 0, stream>>>(
        x, NN, NODE_F, wf + FR_PREN * FRAG_USH, preN_b, out_x, xbf);
    wmma_rowgemm_relu<2><<<NE / 16 / 8, 256, 0, stream>>>(
        edge_attr, NE, EDGE_F, wf + FR_PREE * FRAG_USH, preE_b, out_e, ebf);

    // ---- layers ----
    for (int i = 0; i < NL; ++i) {
        zero_buf<<<2, 256, 0, stream>>>(stats, S_TOTAL);
        node_linear<<<(NN / 16 + 7) / 8, 256, 0, stream>>>(
            xbf, wf + (FR_SRC + i * 8) * FRAG_USH, wf + (FR_DST + i * 8) * FRAG_USH,
            src_b + i * DIMF, dst_b + i * DIMF, hbuf, dstf);
        edge_gate<<<NE / 16 / 8, 256, 0, stream>>>(
            xbf, ebf, ei_src, ei_dst, wf + (FR_GATE + i * 20) * FRAG_USH,
            gate_b + i * DIMF, dstf, hbuf, mbuf, stats);
        col_stats<<<512, 256, 0, stream>>>(hbuf, NN * DIMF,
                                           stats + S_SUM_N, stats + S_SQ_N);
        bn_finalize<<<1, 64, 0, stream>>>(stats + S_SUM_N, stats + S_SQ_N, 1.0f / NN,
                                          bn_in_n_g + i * DIMF, bn_in_n_b + i * DIMF,
                                          stats + S_SC1N, stats + S_SH1N);
        bn_finalize<<<1, 64, 0, stream>>>(stats + S_SUM_E, stats + S_SQ_E, 1.0f / NE,
                                          bn_in_e_g + i * DIMF, bn_in_e_b + i * DIMF,
                                          stats + S_SC1E, stats + S_SH1E);
        bn_apply_relu_stats<<<512, 256, 0, stream>>>(hbuf, NN * DIMF,
                                                     stats + S_SC1N, stats + S_SH1N,
                                                     stats + S_SUM2N, stats + S_SQ2N);
        bn_apply_relu_stats<<<2048, 256, 0, stream>>>(mbuf, NE * DIMF,
                                                      stats + S_SC1E, stats + S_SH1E,
                                                      stats + S_SUM2E, stats + S_SQ2E);
        bn_finalize<<<1, 64, 0, stream>>>(stats + S_SUM2N, stats + S_SQ2N, 1.0f / NN,
                                          bn_out_n_g + i * DIMF, bn_out_n_b + i * DIMF,
                                          stats + S_SC2N, stats + S_SH2N);
        bn_finalize<<<1, 64, 0, stream>>>(stats + S_SUM2E, stats + S_SQ2E, 1.0f / NE,
                                          bn_out_e_g + i * DIMF, bn_out_e_b + i * DIMF,
                                          stats + S_SC2E, stats + S_SH2E);
        bn_residual_mirror<<<512, 256, 0, stream>>>(hbuf, NN * DIMF,
                                                    stats + S_SC2N, stats + S_SH2N,
                                                    out_x, xbf);
        bn_residual_mirror<<<2048, 256, 0, stream>>>(mbuf, NE * DIMF,
                                                     stats + S_SC2E, stats + S_SH2E,
                                                     out_e, ebf);
    }

    // ---- readout ----
    zero_buf<<<4, 256, 0, stream>>>(pooled, NG * DIMF);
    pool_kernel<<<512, 256, 0, stream>>>(out_x, batch, pooled);
    head_kernel<<<1, 256, 0, stream>>>(pooled, post_W, post_b, out_W, out_b,
                                       (float*)d_out);
}
